// RelationNet_9002251452554
// MI455X (gfx1250) — compile-verified, matrix operand order checked by reading, not run on hardware
//
#include <hip/hip_runtime.h>
#include <stdint.h>

// ---------------- CDNA5 WMMA types ----------------
typedef __attribute__((ext_vector_type(16))) __bf16 v16bf;
typedef __attribute__((ext_vector_type(8)))  float  v8f;

#define DEV __device__ __forceinline__

// ---------------- problem constants ----------------
static constexpr int FEAT = 2048, HW = 14, C = 20, M = 64, U = 128, B = 4;
static constexpr int HW2 = HW * HW;        // 196
static constexpr int CM  = C * M;          // 1280 (rows of GEMM1)
static constexpr int MT1 = CM / 16;        // 80 M-tiles, GEMM1
static constexpr int KT1 = FEAT / 32;      // 64 K-steps, GEMM1
static constexpr int NT  = (HW2 + 15) / 16;// 13 N-tiles (196 -> 208 padded)
static constexpr int NTP = (NT + 1) / 2;   // 7 N-tile pairs per wave, GEMM1
static constexpr int UT  = U / 16;         // 8 U-tiles, GEMM2
static constexpr int KS2 = (2 * M) / 32;   // 4 K-steps, GEMM2
static constexpr int FR  = 512;            // bf16 elems per fragment (32 lanes x 16)

// ---------------- workspace layout (bytes) ----------------
static constexpr size_t OFF_WPF = 0;                                   // W_parcel A-frags
static constexpr size_t SZ_WPF  = (size_t)MT1 * KT1 * FR * 2;          // 5,242,880
static constexpr size_t OFF_XWF = OFF_WPF + SZ_WPF;                    // x B-frags
static constexpr size_t SZ_XWF  = (size_t)B * KT1 * NT * FR * 2;       // 3,407,872
static constexpr size_t OFF_WGF = OFF_XWF + SZ_XWF;                    // Wg A-frags
static constexpr size_t SZ_WGF  = (size_t)C * (C - 1) * UT * KS2 * FR * 2;
static constexpr size_t OFF_XOF = OFF_WGF + SZ_WGF;                    // x_obj B-frags
static constexpr size_t SZ_XOF  = (size_t)B * C * 2 * NT * FR * 2;
static constexpr size_t OFF_XT  = OFF_XOF + SZ_XOF;                    // x_tmp f32
static constexpr size_t SZ_XT   = (size_t)B * CM * HW2 * 4;
static constexpr size_t OFF_XO  = OFF_XT + SZ_XT;                      // x_obj f32
static constexpr size_t SZ_XO   = SZ_XT;
static constexpr size_t OFF_REL = OFF_XO + SZ_XO;                      // rel row sums
// total ~31.3 MB

// ---------------- helpers ----------------
DEV uint16_t f2bf(float f) {                 // f32 -> bf16, round-to-nearest-even
  uint32_t u = __float_as_uint(f);
  u += 0x7fffu + ((u >> 16) & 1u);
  return (uint16_t)(u >> 16);
}

// Fragment element -> (row/col, K) per ISA 7.12.2 (wave32, 16-bit A/B 16x32):
// lanes 0-15: VGPR0-3 hold K 0..7, VGPR4-7 hold K 16..23; lanes 16-31: +8.
DEV void frag_coord(int lane, int e, int& mn, int& k) {
  int base = (lane < 16) ? 0 : 8;
  mn = lane & 15;
  int r = e >> 1, h = e & 1;
  k = base + ((r < 4) ? (2 * r + h) : (16 + 2 * (r - 4) + h));
}

DEV v16bf ldfrag(const uint16_t* p) { return *(const v16bf*)p; }

// ---------------- packing kernels ----------------
// A-matrix (rows x Kcols), row-major, -> fragment-major bf16 tiles (mt,kt)
__global__ void pack_a_bf16(const float* __restrict__ src, uint16_t* __restrict__ dst,
                            int nMat, long matStride, int R, int Kc, int ld,
                            int Mt, int Kt) {
  long total = (long)nMat * Mt * Kt * FR;
  long idx = (long)blockIdx.x * blockDim.x + threadIdx.x;
  if (idx >= total) return;
  int e = idx & 15;
  int lane = (int)((idx >> 4) & 31);
  long f = idx >> 9;
  int kt = (int)(f % Kt);
  long t = f / Kt;
  int mt = (int)(t % Mt);
  int mat = (int)(t / Mt);
  int mn, k; frag_coord(lane, e, mn, k);
  int row = mt * 16 + mn, col = kt * 32 + k;
  float v = (row < R && col < Kc) ? src[(size_t)mat * matStride + (size_t)row * ld + col] : 0.f;
  dst[idx] = f2bf(v);
}

// B-matrix (K x N), row-major, -> fragment-major bf16 tiles (kt,nt)
__global__ void pack_b_bf16(const float* __restrict__ src, uint16_t* __restrict__ dst,
                            int nMat, long matStride, int K, int N,
                            int Kt, int Nt) {
  long total = (long)nMat * Kt * Nt * FR;
  long idx = (long)blockIdx.x * blockDim.x + threadIdx.x;
  if (idx >= total) return;
  int e = idx & 15;
  int lane = (int)((idx >> 4) & 31);
  long f = idx >> 9;
  int nt = (int)(f % Nt);
  long t = f / Nt;
  int kt = (int)(t % Kt);
  int mat = (int)(t / Kt);
  int mn, k; frag_coord(lane, e, mn, k);
  int row = kt * 32 + k, col = nt * 16 + mn;
  float v = (row < K && col < N) ? src[(size_t)mat * matStride + (size_t)row * N + col] : 0.f;
  dst[idx] = f2bf(v);
}

// ---------------- GEMM1: x_tmp = W_parcel (1280x2048) @ x_b (2048x196) ----------------
// One wave owns one 16-row M-tile and TWO 16-col N-tiles: each A fragment
// loaded once feeds two v_wmma (2x math per A byte).
__global__ void gemm1_wmma(const uint16_t* __restrict__ WpF,
                           const uint16_t* __restrict__ XwF,
                           float* __restrict__ x_tmp) {
  int lane = threadIdx.x & 31;
  int wave = (int)((blockIdx.x * blockDim.x + threadIdx.x) >> 5);
  int ntp = wave % NTP;
  int mt  = (wave / NTP) % MT1;
  int b   = wave / (NTP * MT1);
  int nt0 = ntp * 2;
  int nt1 = nt0 + 1;
  int nt1c = (nt1 < NT) ? nt1 : nt0;         // clamp: duplicate work, store-masked

  const uint16_t* ap  = WpF + (size_t)(mt * KT1) * FR + lane * 16;
  const uint16_t* bp0 = XwF + ((size_t)(b * KT1) * NT + nt0)  * FR + lane * 16;
  const uint16_t* bp1 = XwF + ((size_t)(b * KT1) * NT + nt1c) * FR + lane * 16;

  v8f acc0 = {}, acc1 = {};
  for (int kt = 0; kt < KT1; ++kt) {
    v16bf a  = ldfrag(ap);
    v16bf b0 = ldfrag(bp0);
    v16bf b1 = ldfrag(bp1);
    acc0 = __builtin_amdgcn_wmma_f32_16x16x32_bf16(false, a, false, b0,
                                                   (short)0, acc0, false, false);
    acc1 = __builtin_amdgcn_wmma_f32_16x16x32_bf16(false, a, false, b1,
                                                   (short)0, acc1, false, false);
    ap  += FR;
    bp0 += (size_t)NT * FR;
    bp1 += (size_t)NT * FR;
    __builtin_prefetch(ap, 0, 3);            // near-cache speculative prefetch
    __builtin_prefetch(bp0, 0, 3);
  }
  // D fragment: VGPR r -> M = r (lanes 0-15) / r+8 (lanes 16-31), N = lane&15
  int mofs = (lane < 16) ? 0 : 8;
  float* orow = x_tmp + ((size_t)b * CM + (size_t)mt * 16 + mofs) * HW2;
  int col0 = nt0 * 16 + (lane & 15);
  int col1 = nt1 * 16 + (lane & 15);
  if (col0 < HW2) {
#pragma unroll
    for (int r = 0; r < 8; ++r) orow[(size_t)r * HW2 + col0] = acc0[r];
  }
  if (col1 < HW2) {
#pragma unroll
    for (int r = 0; r < 8; ++r) orow[(size_t)r * HW2 + col1] = acc1[r];
  }
}

// ---------------- fused relu-maxpool + STN theta + bilinear grid-sample ----------------
__global__ void pool_stn_sample(const float* __restrict__ x_tmp,
                                const float* __restrict__ W_stn,
                                const float* __restrict__ b_stn,
                                float* __restrict__ x_obj) {
  __shared__ float pooled[M * 49];
  __shared__ float red[256];
  __shared__ float theta[6];
  int bc = blockIdx.x, b = bc / C, c = bc % C;
  int tid = threadIdx.x;
  const float* xt = x_tmp + ((size_t)b * CM + (size_t)c * M) * HW2;

  for (int i = tid; i < M * 49; i += 256) {
    int m = i / 49, p = i % 49, ph = p / 7, pw = p % 7;
    const float* base = xt + (size_t)m * HW2 + (2 * ph) * HW + 2 * pw;
    float v = fmaxf(fmaxf(base[0], base[1]), fmaxf(base[HW], base[HW + 1]));
    pooled[i] = fmaxf(v, 0.f);           // relu folded into pool (max with 0)
  }
  __syncthreads();

  for (int j = 0; j < 6; ++j) {
    float p = 0.f;
    const float* wsrow = W_stn + ((size_t)c * 6 + j) * (M * 49);
    for (int k = tid; k < M * 49; k += 256) p += pooled[k] * wsrow[k];
    red[tid] = p; __syncthreads();
    for (int s = 128; s > 0; s >>= 1) { if (tid < s) red[tid] += red[tid + s]; __syncthreads(); }
    if (tid == 0) theta[j] = red[0] + b_stn[c * 6 + j];
    __syncthreads();
  }

  float* xo = x_obj + ((size_t)b * CM + (size_t)c * M) * HW2;
  for (int i = tid; i < M * HW2; i += 256) {
    int m = i / HW2, p = i % HW2, h = p / HW, w = p % HW;
    float xs = -1.f + 2.f * w / (HW - 1);
    float ys = -1.f + 2.f * h / (HW - 1);
    float gx = ((theta[0] * xs + theta[1] * ys + theta[2]) + 1.f) * 0.5f * (HW - 1);
    float gy = ((theta[3] * xs + theta[4] * ys + theta[5]) + 1.f) * 0.5f * (HW - 1);
    int x0 = (int)floorf(gx), y0 = (int)floorf(gy), x1 = x0 + 1, y1 = y0 + 1;
    const float* img = xt + (size_t)m * HW2;
    auto gat = [&](int yi, int xi) -> float {
      bool valid = (yi >= 0) && (yi < HW) && (xi >= 0) && (xi < HW);
      int yc = min(max(yi, 0), HW - 1), xc = min(max(xi, 0), HW - 1);
      return valid ? img[yc * HW + xc] : 0.f;
    };
    float w00 = (x1 - gx) * (y1 - gy), w01 = (gx - x0) * (y1 - gy);
    float w10 = (x1 - gx) * (gy - y0), w11 = (gx - x0) * (gy - y0);
    xo[i] = gat(y0, x0) * w00 + gat(y0, x1) * w01 + gat(y1, x0) * w10 + gat(y1, x1) * w11;
  }
}

// ---------------- GEMM2: rel_sum[b,c,u] = sum_{k,hw} relu(Wg[c,k] @ [xo_c; xo_jidx]) ----------------
// k outer / nt inner: the 4 Wg A-fragments for (c,k,ut) stay in registers
// (32 VGPRs) and are reused by all 13 N-tiles -> inner loop is 4 B-loads + 4 WMMAs.
__global__ void gemm2_rel(const uint16_t* __restrict__ WgF,
                          const uint16_t* __restrict__ XoF,
                          float* __restrict__ rel_sum) {
  int bc = blockIdx.x, b = bc / C, c = bc % C;
  int lane = threadIdx.x & 31;
  int ut = threadIdx.x >> 5;                  // 8 waves = 8 U-tiles
  float rowsum[8] = {0.f, 0.f, 0.f, 0.f, 0.f, 0.f, 0.f, 0.f};
  bool colOK_base = true;

  for (int k = 0; k < C - 1; ++k) {
    int ch2 = k + (k >= c ? 1 : 0);           // jidx[c][k]
    const uint16_t* abase =
        WgF + ((size_t)(((c * (C - 1) + k) * UT + ut) * KS2)) * FR + lane * 16;
    v16bf a0 = ldfrag(abase + 0 * FR);        // Wa cols  0..31
    v16bf a1 = ldfrag(abase + 1 * FR);        // Wa cols 32..63
    v16bf a2 = ldfrag(abase + 2 * FR);        // Wb cols  0..31
    v16bf a3 = ldfrag(abase + 3 * FR);        // Wb cols 32..63
    const uint16_t* bcs  = XoF + ((size_t)((b * C + c)   * 2) * NT) * FR + lane * 16;
    const uint16_t* bch2 = XoF + ((size_t)((b * C + ch2) * 2) * NT) * FR + lane * 16;

    for (int nt = 0; nt < NT; ++nt) {
      v8f acc = {};
      acc = __builtin_amdgcn_wmma_f32_16x16x32_bf16(false, a0, false,
              ldfrag(bcs  + (size_t)(0 * NT + nt) * FR), (short)0, acc, false, false);
      acc = __builtin_amdgcn_wmma_f32_16x16x32_bf16(false, a1, false,
              ldfrag(bcs  + (size_t)(1 * NT + nt) * FR), (short)0, acc, false, false);
      acc = __builtin_amdgcn_wmma_f32_16x16x32_bf16(false, a2, false,
              ldfrag(bch2 + (size_t)(0 * NT + nt) * FR), (short)0, acc, false, false);
      acc = __builtin_amdgcn_wmma_f32_16x16x32_bf16(false, a3, false,
              ldfrag(bch2 + (size_t)(1 * NT + nt) * FR), (short)0, acc, false, false);
      if ((nt * 16 + (lane & 15)) < HW2 && colOK_base) {
#pragma unroll
        for (int r = 0; r < 8; ++r) rowsum[r] += fmaxf(acc[r], 0.f);
      }
    }
  }
  // reduce over the 16 lanes of each half (each holds one N column of this row)
#pragma unroll
  for (int r = 0; r < 8; ++r) {
    float v = rowsum[r];
    v += __shfl_xor(v, 1, 32);
    v += __shfl_xor(v, 2, 32);
    v += __shfl_xor(v, 4, 32);
    v += __shfl_xor(v, 8, 32);
    rowsum[r] = v;
  }
  if ((lane & 15) == 0) {
    int mofs = (lane < 16) ? 0 : 8;
#pragma unroll
    for (int r = 0; r < 8; ++r)
      rel_sum[(size_t)bc * U + ut * 16 + r + mofs] = rowsum[r];
  }
}

// ---------------- finalize: out[b,c] = mean_u-dot(w_phi) + b_phi ----------------
__global__ void finalize(const float* __restrict__ rel_sum,
                         const float* __restrict__ w_phi,
                         const float* __restrict__ b_phi,
                         float* __restrict__ out) {
  __shared__ float red[128];
  int bc = blockIdx.x, tid = threadIdx.x;
  red[tid] = rel_sum[(size_t)bc * U + tid] * (1.0f / (float)HW2) * w_phi[tid];
  __syncthreads();
  for (int s = 64; s > 0; s >>= 1) { if (tid < s) red[tid] += red[tid + s]; __syncthreads(); }
  if (tid == 0) out[bc] = red[0] + b_phi[0];
}

// ---------------- host launch ----------------
extern "C" void kernel_launch(void* const* d_in, const int* in_sizes, int n_in,
                              void* d_out, int out_size, void* d_ws, size_t ws_size,
                              hipStream_t stream) {
  const float* x        = (const float*)d_in[0];
  const float* W_parcel = (const float*)d_in[1];
  const float* W_stn    = (const float*)d_in[2];
  const float* b_stn    = (const float*)d_in[3];
  const float* Wg       = (const float*)d_in[4];
  const float* w_phi    = (const float*)d_in[5];
  const float* b_phi    = (const float*)d_in[6];
  float* out = (float*)d_out;

  char* ws = (char*)d_ws;
  uint16_t* WpF = (uint16_t*)(ws + OFF_WPF);
  uint16_t* XwF = (uint16_t*)(ws + OFF_XWF);
  uint16_t* WgF = (uint16_t*)(ws + OFF_WGF);
  uint16_t* XoF = (uint16_t*)(ws + OFF_XOF);
  float* x_tmp   = (float*)(ws + OFF_XT);
  float* x_obj   = (float*)(ws + OFF_XO);
  float* rel_sum = (float*)(ws + OFF_REL);

  // pack W_parcel (1280 x 2048) as A-fragments
  {
    long n = (long)MT1 * KT1 * FR;
    pack_a_bf16<<<(int)((n + 255) / 256), 256, 0, stream>>>(
        W_parcel, WpF, 1, 0, CM, FEAT, FEAT, MT1, KT1);
  }
  // pack x (B x 2048 x 196) as B-fragments
  {
    long n = (long)B * KT1 * NT * FR;
    pack_b_bf16<<<(int)((n + 255) / 256), 256, 0, stream>>>(
        x, XwF, B, (long)FEAT * HW2, FEAT, HW2, KT1, NT);
  }
  // pack Wg (380 matrices of 128 x 128) as A-fragments
  {
    long n = (long)C * (C - 1) * UT * KS2 * FR;
    pack_a_bf16<<<(int)((n + 255) / 256), 256, 0, stream>>>(
        Wg, WgF, C * (C - 1), (long)U * 2 * M, U, 2 * M, 2 * M, UT, KS2);
  }
  // GEMM1: 2240 waves (one per M-tile x N-tile-pair), 8 waves per block
  gemm1_wmma<<<(B * MT1 * NTP) / 8, 256, 0, stream>>>(WpF, XwF, x_tmp);
  // pooled -> theta -> grid sample
  pool_stn_sample<<<B * C, 256, 0, stream>>>(x_tmp, W_stn, b_stn, x_obj);
  // pack x_obj (80 matrices of 64 x 196) as B-fragments
  {
    long n = (long)B * C * 2 * NT * FR;
    pack_b_bf16<<<(int)((n + 255) / 256), 256, 0, stream>>>(
        x_obj, XoF, B * C, (long)M * HW2, M, HW2, 2, NT);
  }
  // GEMM2 + relu + k-sum + hw-sum
  gemm2_rel<<<B * C, 256, 0, stream>>>(WgF, XoF, rel_sum);
  // output projection
  finalize<<<B * C, 128, 0, stream>>>(rel_sum, w_phi, b_phi, out);
}
